// TopicGroupedAttention_23149873726004
// MI455X (gfx1250) — compile-verified, matrix operand order checked by reading, not run on hardware
//
#include <hip/hip_runtime.h>
#include <hip/hip_bf16.h>
#include <math.h>

// Problem constants (match reference)
#define HDIM  1024
#define SEQ   2048
#define BATCH 2
#define NHEAD 16
#define HD    64
#define MROWS (BATCH*SEQ)   // 4096

typedef __attribute__((ext_vector_type(16))) _Float16 v16h;
typedef __attribute__((ext_vector_type(8)))  _Float16 v8h;
typedef __attribute__((ext_vector_type(8)))  float    v8f;

static __device__ __forceinline__ v16h cat8(v8h lo, v8h hi) {
  return __builtin_shufflevector(lo, hi, 0,1,2,3,4,5,6,7,8,9,10,11,12,13,14,15);
}

// Load a 16x32 f16 WMMA fragment from row-major memory (global or LDS).
// Caller passes p = row_base + half*8 (half = lane>>4). ISA K-packing makes the
// fragment two contiguous 16-byte runs -> two b128 loads.
static __device__ __forceinline__ v16h load_frag_rm(const _Float16* p) {
  v8h lo = *(const v8h*)(p);
  v8h hi = *(const v8h*)(p + 16);
  return cat8(lo, hi);
}

static __device__ __forceinline__ v8f wmma16(v16h a, v16h b, v8f c) {
  return __builtin_amdgcn_wmma_f32_16x16x32_f16(false, a, false, b, (short)0, c,
                                                false, false);
}

// CDNA5 async copy: global -> LDS, 16 bytes per lane, tracked by ASYNCcnt.
static __device__ __forceinline__ void async_b128(uint32_t lds_off,
                                                  const void* gptr) {
  asm volatile("global_load_async_to_lds_b128 %0, %1, off"
               :: "v"(lds_off), "v"((uint64_t)(uintptr_t)gptr)
               : "memory");
}
static __device__ __forceinline__ uint32_t lds_off_of(const void* p) {
  return (uint32_t)(uintptr_t)p;   // generic LDS pointer: low 32 bits = offset
}

// ---------------------------------------------------------------------------
// 1) elementwise f32 -> f16
__global__ void cvt_f32_to_f16(const float* __restrict__ src,
                               _Float16* __restrict__ dst, int n) {
  int i = blockIdx.x * blockDim.x + threadIdx.x;
  if (i < n) dst[i] = (_Float16)src[i];
}

// ---------------------------------------------------------------------------
// 2) pack weights [K=1024][N=1024] row-major f32 into WMMA B-fragment tiles:
//    dst index = ((kt*64 + nt)*32 + lane)*16 halfs  (32 bytes contiguous/lane)
__global__ void pack_w(const float* __restrict__ W0, const float* __restrict__ W1,
                       const float* __restrict__ W2, const float* __restrict__ W3,
                       _Float16* __restrict__ P0, _Float16* __restrict__ P1,
                       _Float16* __restrict__ P2, _Float16* __restrict__ P3) {
  int t = blockIdx.x * blockDim.x + threadIdx.x;   // 4*32*64*32 = 262144 threads
  int lane = t & 31; t >>= 5;
  int nt   = t & 63; t >>= 6;
  int kt   = t & 31; t >>= 5;
  int w    = t;
  const float* W = (w == 0) ? W0 : (w == 1) ? W1 : (w == 2) ? W2 : W3;
  _Float16*    P = (w == 0) ? P0 : (w == 1) ? P1 : (w == 2) ? P2 : P3;
  int n    = nt * 16 + (lane & 15);
  int half = lane >> 4;
  v16h frag = {};
#pragma unroll
  for (int v = 0; v < 8; ++v) {
    int kb = kt * 32 + (v >> 2) * 16 + half * 8 + (v & 3) * 2;
    frag[2 * v]     = (_Float16)W[(size_t)(kb    ) * HDIM + n];
    frag[2 * v + 1] = (_Float16)W[(size_t)(kb + 1) * HDIM + n];
  }
  *(v16h*)(P + ((size_t)((kt * 64 + nt) * 32 + lane)) * 16) = frag;
}

// ---------------------------------------------------------------------------
// 3/4) tiny GEMV for the topic gate chain (B=2 rows only)
__global__ void gemv_bias(const float* __restrict__ x, const float* __restrict__ W,
                          const float* __restrict__ bias, float* __restrict__ out,
                          int sig) {
  int idx = blockIdx.x * blockDim.x + threadIdx.x;
  if (idx >= BATCH * HDIM) return;
  int b = idx / HDIM, j = idx % HDIM;
  const float* xr = x + (size_t)b * HDIM;
  float s = bias[j];
  for (int i = 0; i < HDIM; ++i) s += xr[i] * W[(size_t)i * HDIM + j];
  out[idx] = sig ? (1.0f / (1.0f + __expf(-s))) : s;
}

// ---------------------------------------------------------------------------
// 5) fused QKV projection. Block = 8 waves, all on the SAME 64-col group (ng)
//    over 8 row tiles, so the 4KB weight panel per K-step is async-staged into
//    LDS once per block (double buffered), and A-frags are register
//    double-buffered. q,k row-major f16; v transposed [B][H][S] f16.
__global__ void __launch_bounds__(256) qkv_gemm(
    const _Float16* __restrict__ hs16,
    const _Float16* __restrict__ WqP, const _Float16* __restrict__ WkP,
    const _Float16* __restrict__ WvP,
    const float* __restrict__ bq, const float* __restrict__ bk,
    const float* __restrict__ bv, const float* __restrict__ gate,
    _Float16* __restrict__ q16, _Float16* __restrict__ k16,
    _Float16* __restrict__ vt16) {
  __shared__ __align__(16) _Float16 bstage[2][2048];     // 2 x 4KB B panels
  const int tid = threadIdx.x;
  const int lane = tid & 31, wslot = tid >> 5;
  int x = blockIdx.x;                  // 1536 = 3 mats * 32 mtg * 16 ng
  const int mat = x / 512; x -= mat * 512;
  const int mtg = x >> 4;
  const int ng  = x & 15;
  const int mt  = mtg * 8 + wslot;     // 0..255
  const _Float16* WP = (mat == 0) ? WqP : (mat == 1) ? WkP : WvP;
  const int half = lane >> 4, ln16 = lane & 15;

  const char* wsrc = (const char*)WP + (size_t)(ng * 4) * 1024 + tid * 16;
  auto issue = [&](int kt, int s) {    // stage one 4KB panel: 256 thr x 16B
    async_b128(lds_off_of(&bstage[s][0]) + tid * 16,
               wsrc + (size_t)kt * 64 * 1024);
  };

  v8f acc0 = {}, acc1 = {}, acc2 = {}, acc3 = {};
  const _Float16* arow = hs16 + (size_t)(mt * 16 + ln16) * HDIM + half * 8;
  issue(0, 0);
  v16h a_cur = load_frag_rm(arow);
  for (int kt = 0; kt < 32; ++kt) {
    const int s = kt & 1;
    if (kt + 1 < 32) {
      issue(kt + 1, s ^ 1);
      asm volatile("s_wait_asynccnt 0x1" ::: "memory");
    } else {
      asm volatile("s_wait_asynccnt 0x0" ::: "memory");
    }
    __syncthreads();                                   // panel kt visible
    v16h a_next = (kt + 1 < 32) ? load_frag_rm(arow + (kt + 1) * 32) : a_cur;
    const _Float16* bp = &bstage[s][(size_t)lane * 16];
    v16h b0 = *(const v16h*)(bp);
    v16h b1 = *(const v16h*)(bp + 512);
    v16h b2 = *(const v16h*)(bp + 1024);
    v16h b3 = *(const v16h*)(bp + 1536);
    acc0 = wmma16(a_cur, b0, acc0);
    acc1 = wmma16(a_cur, b1, acc1);
    acc2 = wmma16(a_cur, b2, acc2);
    acc3 = wmma16(a_cur, b3, acc3);
    a_cur = a_next;
    __syncthreads();                                   // reads done before reuse
  }
  v8f accs[4] = {acc0, acc1, acc2, acc3};
  const int bb = (mt * 16) >> 11;
#pragma unroll
  for (int i = 0; i < 4; ++i) {
    int c = ng * 64 + i * 16 + ln16;
    float bias = (mat == 0 ? bq : (mat == 1 ? bk : bv))[c];
    float g = (mat == 1) ? gate[(size_t)bb * HDIM + c] : 1.0f;
#pragma unroll
    for (int j = 0; j < 8; ++j) {
      int r = mt * 16 + j + half * 8;
      float val = (accs[i][j] + bias) * ((mat == 1) ? g : 1.0f);
      if (mat == 0)      q16[(size_t)r * HDIM + c] = (_Float16)val;
      else if (mat == 1) k16[(size_t)r * HDIM + c] = (_Float16)val;
      else {
        int ss = r & (SEQ - 1);
        vt16[((size_t)bb * HDIM + c) * SEQ + ss] = (_Float16)val;
      }
    }
  }
}

// ---------------------------------------------------------------------------
// 6) flash attention. Block = 8 waves on 8 query tiles of the SAME (b,h):
//    per 32-key step, K panel (32x64 f16) and V panel (64x32 f16) are
//    async-staged into LDS once per block, double buffered. Online softmax;
//    P transposed D->A layout through per-wave LDS.
__global__ void __launch_bounds__(256) attn_kernel(
    const _Float16* __restrict__ q16, const _Float16* __restrict__ k16,
    const _Float16* __restrict__ vt16, const float* __restrict__ mask,
    _Float16* __restrict__ ctx16) {
  __shared__ __align__(16) _Float16 kst[2][32][64];    // 2 x 4KB key panels
  __shared__ __align__(16) _Float16 vst[2][64][32];    // 2 x 4KB value panels
  __shared__ _Float16 pbuf[8][16][32];                 // 1KB per wave
  const int tid = threadIdx.x;
  const int lane = tid & 31, wslot = tid >> 5;
  int x = blockIdx.x;                  // 512 = B(2) * NH(16) * 16 qt-groups
  const int qtg = x & 15;
  const int h   = (x >> 4) & 15;
  const int b   = x >> 8;
  const int qt  = qtg * 8 + wslot;     // 0..127
  const int half = lane >> 4, ln16 = lane & 15;

  const _Float16* qrow = q16 + (size_t)(b * SEQ + qt * 16 + ln16) * HDIM + h * HD + half * 8;
  v16h aq0 = load_frag_rm(qrow);
  v16h aq1 = load_frag_rm(qrow + 32);
  const char* kbase = (const char*)(k16 + (size_t)(b * SEQ) * HDIM + h * HD);
  const char* vbase = (const char*)(vt16 + ((size_t)b * HDIM + h * HD) * SEQ);
  const float* mrow = mask + (size_t)b * SEQ;

  // Stage K (32 rows x 128B, row stride HDIM*2) and V (64 rows x 64B, row
  // stride SEQ*2) for key block kk. Linear LDS offset == tid*16 by layout.
  auto issue = [&](int kk, int s) {
    { int row = tid >> 3, ch = tid & 7;
      async_b128(lds_off_of(&kst[s][0][0]) + tid * 16,
                 kbase + (size_t)(kk + row) * (HDIM * 2) + ch * 16); }
    { int row = tid >> 2, ch = tid & 3;
      async_b128(lds_off_of(&vst[s][0][0]) + tid * 16,
                 vbase + (size_t)row * (SEQ * 2) + (size_t)kk * 2 + ch * 16); }
  };

  float mrun[8], lrun[8];
#pragma unroll
  for (int j = 0; j < 8; ++j) { mrun[j] = -1e30f; lrun[j] = 0.0f; }
  v8f acc0 = {}, acc1 = {}, acc2 = {}, acc3 = {};

  issue(0, 0);
  for (int kkI = 0; kkI < SEQ / 32; ++kkI) {
    const int kk = kkI * 32, s = kkI & 1;
    if (kkI + 1 < SEQ / 32) {
      issue(kk + 32, s ^ 1);
      asm volatile("s_wait_asynccnt 0x2" ::: "memory");
    } else {
      asm volatile("s_wait_asynccnt 0x0" ::: "memory");
    }
    __syncthreads();                                   // panels kk visible

    v8f s0 = {}, s1 = {};
    s0 = wmma16(aq0, load_frag_rm(&kst[s][ln16][half * 8]),           s0);
    s0 = wmma16(aq1, load_frag_rm(&kst[s][ln16][32 + half * 8]),      s0);
    s1 = wmma16(aq0, load_frag_rm(&kst[s][16 + ln16][half * 8]),      s1);
    s1 = wmma16(aq1, load_frag_rm(&kst[s][16 + ln16][32 + half * 8]), s1);
    float msk0 = mrow[kk + ln16];
    float msk1 = mrow[kk + 16 + ln16];
#pragma unroll
    for (int j = 0; j < 8; ++j) {
      s0[j] = s0[j] * 0.125f + msk0;                   // 1/sqrt(64)
      s1[j] = s1[j] * 0.125f + msk1;
    }
    float corr[8];
#pragma unroll
    for (int j = 0; j < 8; ++j) {
      float t = fmaxf(s0[j], s1[j]);
#pragma unroll
      for (int o = 8; o; o >>= 1) t = fmaxf(t, __shfl_xor(t, o, 16));
      float mn = fmaxf(mrun[j], t);
      corr[j] = __expf(mrun[j] - mn);
      mrun[j] = mn;
      float p0 = __expf(s0[j] - mn);
      float p1 = __expf(s1[j] - mn);
      s0[j] = p0; s1[j] = p1;
      float r = p0 + p1;
#pragma unroll
      for (int o = 8; o; o >>= 1) r += __shfl_xor(r, o, 16);
      lrun[j] = lrun[j] * corr[j] + r;
    }
#pragma unroll
    for (int j = 0; j < 8; ++j) {
      acc0[j] *= corr[j]; acc1[j] *= corr[j];
      acc2[j] *= corr[j]; acc3[j] *= corr[j];
    }
    // P: D-layout -> LDS -> A-layout (same-wave DS ops in order)
#pragma unroll
    for (int j = 0; j < 8; ++j) {
      int m = j + half * 8;
      pbuf[wslot][m][ln16]      = (_Float16)s0[j];
      pbuf[wslot][m][16 + ln16] = (_Float16)s1[j];
    }
    v16h ap = load_frag_rm(&pbuf[wslot][ln16][half * 8]);
    acc0 = wmma16(ap, load_frag_rm(&vst[s][ln16][half * 8]),      acc0);
    acc1 = wmma16(ap, load_frag_rm(&vst[s][16 + ln16][half * 8]), acc1);
    acc2 = wmma16(ap, load_frag_rm(&vst[s][32 + ln16][half * 8]), acc2);
    acc3 = wmma16(ap, load_frag_rm(&vst[s][48 + ln16][half * 8]), acc3);
    __syncthreads();                                   // reads done before reuse
  }
#pragma unroll
  for (int j = 0; j < 8; ++j) {
    float inv = 1.0f / lrun[j];
    size_t row = (size_t)(b * SEQ + qt * 16 + j + half * 8) * HDIM + h * HD;
    ctx16[row + ln16]      = (_Float16)(acc0[j] * inv);
    ctx16[row + 16 + ln16] = (_Float16)(acc1[j] * inv);
    ctx16[row + 32 + ln16] = (_Float16)(acc2[j] * inv);
    ctx16[row + 48 + ln16] = (_Float16)(acc3[j] * inv);
  }
}

// ---------------------------------------------------------------------------
// 7) output projection + bias + residual, same async B-panel staging.
__global__ void __launch_bounds__(256) out_gemm(
    const _Float16* __restrict__ ctx16, const _Float16* __restrict__ WdP,
    const float* __restrict__ bd, const float* __restrict__ hidden,
    float* __restrict__ xbuf) {
  __shared__ __align__(16) _Float16 bstage[2][2048];
  const int tid = threadIdx.x;
  const int lane = tid & 31, wslot = tid >> 5;
  const int mtg = blockIdx.x >> 4;     // 512 blocks = 32 mtg * 16 ng
  const int ng  = blockIdx.x & 15;
  const int mt  = mtg * 8 + wslot;
  const int half = lane >> 4, ln16 = lane & 15;

  const char* wsrc = (const char*)WdP + (size_t)(ng * 4) * 1024 + tid * 16;
  auto issue = [&](int kt, int s) {
    async_b128(lds_off_of(&bstage[s][0]) + tid * 16,
               wsrc + (size_t)kt * 64 * 1024);
  };

  v8f acc0 = {}, acc1 = {}, acc2 = {}, acc3 = {};
  const _Float16* arow = ctx16 + (size_t)(mt * 16 + ln16) * HDIM + half * 8;
  issue(0, 0);
  v16h a_cur = load_frag_rm(arow);
  for (int kt = 0; kt < 32; ++kt) {
    const int s = kt & 1;
    if (kt + 1 < 32) {
      issue(kt + 1, s ^ 1);
      asm volatile("s_wait_asynccnt 0x1" ::: "memory");
    } else {
      asm volatile("s_wait_asynccnt 0x0" ::: "memory");
    }
    __syncthreads();
    v16h a_next = (kt + 1 < 32) ? load_frag_rm(arow + (kt + 1) * 32) : a_cur;
    const _Float16* bp = &bstage[s][(size_t)lane * 16];
    v16h b0 = *(const v16h*)(bp);
    v16h b1 = *(const v16h*)(bp + 512);
    v16h b2 = *(const v16h*)(bp + 1024);
    v16h b3 = *(const v16h*)(bp + 1536);
    acc0 = wmma16(a_cur, b0, acc0);
    acc1 = wmma16(a_cur, b1, acc1);
    acc2 = wmma16(a_cur, b2, acc2);
    acc3 = wmma16(a_cur, b3, acc3);
    a_cur = a_next;
    __syncthreads();
  }
  v8f accs[4] = {acc0, acc1, acc2, acc3};
#pragma unroll
  for (int i = 0; i < 4; ++i) {
    int c = ng * 64 + i * 16 + ln16;
    float bias = bd[c];
#pragma unroll
    for (int j = 0; j < 8; ++j) {
      int r = mt * 16 + j + half * 8;
      xbuf[(size_t)r * HDIM + c] = accs[i][j] + bias + hidden[(size_t)r * HDIM + c];
    }
  }
}

// ---------------------------------------------------------------------------
// 8) row-wise LayerNorm (one block per row of 1024)
__global__ void __launch_bounds__(256) layernorm_k(
    const float* __restrict__ x, const float* __restrict__ g,
    const float* __restrict__ lnb, float* __restrict__ out) {
  const int row = blockIdx.x;
  const float* xr = x + (size_t)row * HDIM;
  float s = 0.0f, s2 = 0.0f;
#pragma unroll
  for (int i = 0; i < 4; ++i) {
    float v = xr[threadIdx.x + i * 256];
    s += v; s2 += v * v;
  }
#pragma unroll
  for (int o = 16; o; o >>= 1) { s += __shfl_xor(s, o, 32); s2 += __shfl_xor(s2, o, 32); }
  __shared__ float rs[8], rs2[8];
  if ((threadIdx.x & 31) == 0) { rs[threadIdx.x >> 5] = s; rs2[threadIdx.x >> 5] = s2; }
  __syncthreads();
  s = 0.0f; s2 = 0.0f;
#pragma unroll
  for (int i = 0; i < 8; ++i) { s += rs[i]; s2 += rs2[i]; }
  float mu   = s * (1.0f / HDIM);
  float var  = s2 * (1.0f / HDIM) - mu * mu;
  float rstd = rsqrtf(var + 1e-5f);
#pragma unroll
  for (int i = 0; i < 4; ++i) {
    int c = threadIdx.x + i * 256;
    out[(size_t)row * HDIM + c] = (xr[c] - mu) * rstd * g[c] + lnb[c];
  }
}

// ---------------------------------------------------------------------------
extern "C" void kernel_launch(void* const* d_in, const int* in_sizes, int n_in,
                              void* d_out, int out_size, void* d_ws, size_t ws_size,
                              hipStream_t stream) {
  (void)in_sizes; (void)n_in; (void)out_size; (void)ws_size;
  const float* hs   = (const float*)d_in[0];
  const float* ts   = (const float*)d_in[1];
  const float* mask = (const float*)d_in[2];
  const float* Wq = (const float*)d_in[3];  const float* bq  = (const float*)d_in[4];
  const float* Wk = (const float*)d_in[5];  const float* bk  = (const float*)d_in[6];
  const float* Wv = (const float*)d_in[7];  const float* bv  = (const float*)d_in[8];
  const float* Wtp= (const float*)d_in[9];  const float* btp = (const float*)d_in[10];
  const float* Wta= (const float*)d_in[11]; const float* bta = (const float*)d_in[12];
  const float* Wd = (const float*)d_in[13]; const float* bd  = (const float*)d_in[14];
  const float* lg = (const float*)d_in[15]; const float* lb  = (const float*)d_in[16];
  float* out = (float*)d_out;

  char* w = (char*)d_ws;
  size_t off = 0;
  auto take = [&](size_t bytes) -> char* {
    char* p = w + off;
    off = (off + bytes + 255) & ~(size_t)255;
    return p;
  };
  _Float16* hs16  = (_Float16*)take((size_t)MROWS * HDIM * 2);   // 8 MB
  _Float16* WqP   = (_Float16*)take((size_t)HDIM * HDIM * 2);    // 2 MB each
  _Float16* WkP   = (_Float16*)take((size_t)HDIM * HDIM * 2);
  _Float16* WvP   = (_Float16*)take((size_t)HDIM * HDIM * 2);
  _Float16* WdP   = (_Float16*)take((size_t)HDIM * HDIM * 2);
  float*    tproj = (float*)take((size_t)BATCH * HDIM * 4);
  float*    gate  = (float*)take((size_t)BATCH * HDIM * 4);
  _Float16* q16   = (_Float16*)take((size_t)MROWS * HDIM * 2);
  _Float16* k16   = (_Float16*)take((size_t)MROWS * HDIM * 2);
  _Float16* vt16  = (_Float16*)take((size_t)MROWS * HDIM * 2);
  _Float16* ctx16 = (_Float16*)take((size_t)MROWS * HDIM * 2);
  float*    xbuf  = (float*)take((size_t)MROWS * HDIM * 4);      // 16 MB

  cvt_f32_to_f16<<<(MROWS * HDIM) / 256, 256, 0, stream>>>(hs, hs16, MROWS * HDIM);
  pack_w<<<1024, 256, 0, stream>>>(Wq, Wk, Wv, Wd, WqP, WkP, WvP, WdP);
  gemv_bias<<<8, 256, 0, stream>>>(ts, Wtp, btp, tproj, 0);
  gemv_bias<<<8, 256, 0, stream>>>(tproj, Wta, bta, gate, 1);
  qkv_gemm<<<1536, 256, 0, stream>>>(hs16, WqP, WkP, WvP, bq, bk, bv, gate,
                                     q16, k16, vt16);
  attn_kernel<<<512, 256, 0, stream>>>(q16, k16, vt16, mask, ctx16);
  out_gemm<<<512, 256, 0, stream>>>(ctx16, WdP, bd, hs, xbuf);
  layernorm_k<<<4096, 256, 0, stream>>>(xbuf, lg, lb, out);
}
